// FSSGruNet_73710228734012
// MI455X (gfx1250) — compile-verified, matrix operand order checked by reading, not run on hardware
//
#include <hip/hip_runtime.h>
#include <hip/hip_bf16.h>
#include <stddef.h>

// ---------------------------------------------------------------------------
// MI455X (gfx1250) implementation of FSSGruNet.
//
// Design: the recurrence is latency-bound (26 GFLOP total, 1.2MB weights),
// so we pipeline the 8 GRU+SortAttn layers across 8 workgroups (one WGP
// each), keep each layer's weights (bf16) + recurrent state (f32) resident
// in ~206KB of LDS, and run every matmul through v_wmma_f32_16x16x32_bf16
// with the documented wave32 fragment layouts. Inter-layer handoff is a
// 4-slot global ring with agent-scope acquire/release flags + s_sleep spin.
// Heads/pool are separate parallel kernels (deterministic reductions).
// ---------------------------------------------------------------------------

typedef __bf16 bf16_t;
typedef __attribute__((ext_vector_type(16))) __bf16 v16bf;
typedef __attribute__((ext_vector_type(8)))  float  v8f;

#define NLAYER 8
#define BATCH  64
#define HID    64
#define LW     512
#define NSLOT  4

// ---- dynamic LDS layout (bytes) -------------------------------------------
#define OFF_WIH  0                      // 192x64 bf16
#define OFF_WHH  24576                  // 192x64 bf16
#define OFF_W1   49152                  // 64x64  bf16
#define OFF_W2   57344                  // 32x32  bf16
#define OFF_W3   59392                  // 32x32  bf16
#define OFF_W4   61440                  // 64x128 bf16
#define OFF_BIH  77824                  // 192 f32
#define OFF_BHH  78592                  // 192 f32
#define OFF_H    79360                  // 64x64 f32 state
#define OFF_INP  95744                  // 64x64 bf16
#define OFF_HBF  103936                 // 64x64 bf16
#define OFF_S0   112128                 // 48KB f32 scratch (gi/sort/a1/o2/o3/o4)
#define OFF_S1   161280                 // 48KB scratch (gh f32 | seBf/A2/A3/A4 bf16)
#define LDS_TOTAL 210432

__device__ __forceinline__ float sigmf(float v) {
    return 1.0f / (1.0f + __expf(-v));
}

// A fragment: 16x32 bf16 tile, M rows striped per ISA layout.
__device__ __forceinline__ v16bf load_frag_a(const bf16_t* A, int lda,
                                             int m0, int k0, int lane) {
    v16bf f;
    const int m  = m0 + (lane & 15);
    const int kb = k0 + ((lane & 16) ? 8 : 0);
    const bf16_t* row = A + m * lda;
#pragma unroll
    for (int v = 0; v < 8; ++v) {
        const int k = kb + ((v < 4) ? (2 * v) : (16 + 2 * (v - 4)));
        f[2 * v]     = row[k];
        f[2 * v + 1] = row[k + 1];
    }
    return f;
}

// B fragment from Bt stored as (N x K) row-major (i.e. weight[out][in]).
__device__ __forceinline__ v16bf load_frag_b(const bf16_t* Bt, int ldb,
                                             int n0, int k0, int lane) {
    v16bf f;
    const int n  = n0 + (lane & 15);
    const int kb = k0 + ((lane & 16) ? 8 : 0);
    const bf16_t* row = Bt + n * ldb;
#pragma unroll
    for (int v = 0; v < 8; ++v) {
        const int k = kb + ((v < 4) ? (2 * v) : (16 + 2 * (v - 4)));
        f[2 * v]     = row[k];
        f[2 * v + 1] = row[k + 1];
    }
    return f;
}

__device__ __forceinline__ void store_frag_c(float* C, int ldc,
                                             int m0, int n0, int lane, v8f c) {
    const int n  = n0 + (lane & 15);
    const int mb = m0 + ((lane & 16) ? 8 : 0);
#pragma unroll
    for (int v = 0; v < 8; ++v) C[(mb + v) * ldc + n] = c[v];
}

// C = A(MxK) @ Bt(NxK)^T + bias, all operands in LDS, tiles over 8 waves.
__device__ __forceinline__ void gemm_bf16(const bf16_t* A, int lda,
                                          const bf16_t* Bt, int ldb,
                                          float* C, int ldc,
                                          const float* bias,
                                          int M, int N, int K,
                                          int wave, int lane) {
    const int tn = N >> 4;
    const int nt = (M >> 4) * tn;
    for (int t = wave; t < nt; t += 8) {
        const int m0 = (t / tn) << 4;
        const int n0 = (t % tn) << 4;
        const float bv = bias ? bias[n0 + (lane & 15)] : 0.0f;
        v8f c;
#pragma unroll
        for (int v = 0; v < 8; ++v) c[v] = bv;
        for (int k0 = 0; k0 < K; k0 += 32) {
            v16bf a = load_frag_a(A, lda, m0, k0, lane);
            v16bf b = load_frag_b(Bt, ldb, n0, k0, lane);
            c = __builtin_amdgcn_wmma_f32_16x16x32_bf16(
                    false, a, false, b, (short)0, c, false, false);
        }
        store_frag_c(C, ldc, m0, n0, lane, c);
    }
}

// ---------------------------------------------------------------------------
// Kernel 0: zero the pipeline flags (d_ws is not re-poisoned between calls;
// this keeps kernel_launch deterministic).
// ---------------------------------------------------------------------------
__global__ void fss_init_flags(int* flags) {
    if (threadIdx.x < 16) flags[threadIdx.x] = 0;
}

// ---------------------------------------------------------------------------
// Kernel 1: 8-stage layer pipeline. blockIdx.x = layer, 256 threads = 8 waves.
// ---------------------------------------------------------------------------
__global__ void __launch_bounds__(256)
fss_pipeline(const float* __restrict__ x,
             const float* __restrict__ h0,
             const float* __restrict__ w_ih,
             const float* __restrict__ w_hh,
             const float* __restrict__ b_ih,
             const float* __restrict__ b_hh,
             const float* __restrict__ aw1,
             const float* __restrict__ aw2,
             const float* __restrict__ aw3,
             const float* __restrict__ aw4,
             float* __restrict__ out_ys,
             float* __restrict__ out_hfin,
             int* __restrict__ flags,
             float* __restrict__ ring) {
    extern __shared__ char smem[];
    bf16_t* sWih = (bf16_t*)(smem + OFF_WIH);
    bf16_t* sWhh = (bf16_t*)(smem + OFF_WHH);
    bf16_t* sW1  = (bf16_t*)(smem + OFF_W1);
    bf16_t* sW2  = (bf16_t*)(smem + OFF_W2);
    bf16_t* sW3  = (bf16_t*)(smem + OFF_W3);
    bf16_t* sW4  = (bf16_t*)(smem + OFF_W4);
    float*  sBih = (float*)(smem + OFF_BIH);
    float*  sBhh = (float*)(smem + OFF_BHH);
    float*  sH   = (float*)(smem + OFF_H);
    bf16_t* sInp = (bf16_t*)(smem + OFF_INP);
    bf16_t* sHbf = (bf16_t*)(smem + OFF_HBF);
    float*  sS0  = (float*)(smem + OFF_S0);    // gi / sort / a1 / o2 / o3 / o4
    float*  sGh  = (float*)(smem + OFF_S1);    // gh (f32 phase)
    bf16_t* sS1b = (bf16_t*)(smem + OFF_S1);   // seBf / A2 / A3 / A4 (bf16 phase)

    const int layer = blockIdx.x;
    const int tid   = threadIdx.x;
    const int wave  = tid >> 5;
    const int lane  = tid & 31;

    // ---- stage weights into LDS (f32 -> bf16) ----
    {
        const float* s;
        s = w_ih + (size_t)layer * 12288;
        for (int i = tid; i < 12288; i += 256) sWih[i] = (bf16_t)s[i];
        s = w_hh + (size_t)layer * 12288;
        for (int i = tid; i < 12288; i += 256) sWhh[i] = (bf16_t)s[i];
        s = aw1 + (size_t)layer * 4096;
        for (int i = tid; i < 4096; i += 256) sW1[i] = (bf16_t)s[i];
        s = aw2 + (size_t)layer * 1024;
        for (int i = tid; i < 1024; i += 256) sW2[i] = (bf16_t)s[i];
        s = aw3 + (size_t)layer * 1024;
        for (int i = tid; i < 1024; i += 256) sW3[i] = (bf16_t)s[i];
        s = aw4 + (size_t)layer * 8192;
        for (int i = tid; i < 8192; i += 256) sW4[i] = (bf16_t)s[i];
        if (tid < 192) {
            sBih[tid] = b_ih[layer * 192 + tid];
            sBhh[tid] = b_hh[layer * 192 + tid];
        }
    }
    // ---- init recurrent state: h[b][j] = h0[layer][0][j] ----
    for (int i = tid; i < 4096; i += 256) sH[i] = h0[layer * HID + (i & 63)];
    __syncthreads();

    int* ready = flags;       // producer progress per layer
    int* done  = flags + 8;   // consumer progress per layer

    for (int t = 0; t < LW; ++t) {
        // ------------------- acquire input tile (64x64 -> bf16) -----------
        if (layer == 0) {
            for (int i = tid; i < 4096; i += 256) {
                const int b = i >> 6, k = i & 63;
                sInp[i] = (bf16_t)x[(size_t)b * 32768 + t * 64 + k];
            }
            if (t + 1 < LW && tid < 64)
                __builtin_prefetch(x + (size_t)tid * 32768 + (t + 1) * 64, 0, 1);
        } else {
            if (tid == 0) {
                while (__hip_atomic_load(&ready[layer - 1], __ATOMIC_ACQUIRE,
                                         __HIP_MEMORY_SCOPE_AGENT) < t + 1)
                    __builtin_amdgcn_s_sleep(1);
            }
            __syncthreads();
            __threadfence();
            const float* slot =
                ring + ((size_t)(layer - 1) * NSLOT + (t & (NSLOT - 1))) * 4096;
            for (int i = tid; i < 4096; i += 256) sInp[i] = (bf16_t)slot[i];
            __syncthreads();
            if (tid == 0)
                __hip_atomic_store(&done[layer - 1], t + 1, __ATOMIC_RELEASE,
                                   __HIP_MEMORY_SCOPE_AGENT);
        }
        for (int i = tid; i < 4096; i += 256) sHbf[i] = (bf16_t)sH[i];
        __syncthreads();

        // ------------------- GRU gates: gi, gh (64x192, K=64) --------------
        gemm_bf16(sInp, 64, sWih, 64, sS0, 192, sBih, 64, 192, 64, wave, lane);
        gemm_bf16(sHbf, 64, sWhh, 64, sGh, 192, sBhh, 64, 192, 64, wave, lane);
        __syncthreads();
        for (int i = tid; i < 4096; i += 256) {
            const int b = i >> 6, j = i & 63, base = b * 192;
            const float r = sigmf(sS0[base + j] + sGh[base + j]);
            const float z = sigmf(sS0[base + 64 + j] + sGh[base + 64 + j]);
            const float n = tanhf(sS0[base + 128 + j] + r * sGh[base + 128 + j]);
            sH[i] = (1.0f - z) * n + z * sH[i];
        }
        __syncthreads();

        // ------------------- SortAttn ---------------------------------------
        // bitonic sort each row of h ascending (into sS0, reused)
        float* srt = sS0;
        for (int i = tid; i < 4096; i += 256) srt[i] = sH[i];
        __syncthreads();
        for (int k = 2; k <= 64; k <<= 1) {
            for (int j = k >> 1; j > 0; j >>= 1) {
                for (int p = tid; p < 2048; p += 256) {
                    const int row = p >> 5, q = p & 31;
                    const int i0 = ((q & ~(j - 1)) << 1) | (q & (j - 1));
                    const int i1 = i0 | j;
                    float* r0 = srt + row * 64;
                    const float a = r0[i0], c = r0[i1];
                    const bool up = ((i0 & k) == 0);
                    if (up ? (a > c) : (c > a)) { r0[i0] = c; r0[i1] = a; }
                }
                __syncthreads();
            }
        }
        // se = stack([h, sorted]) -> (128 x 64) bf16, row r = 2*b + s
        for (int i = tid; i < 8192; i += 256) {
            const int r = i >> 6, col = i & 63, b = r >> 1;
            const float v = (r & 1) ? srt[b * 64 + col] : sH[b * 64 + col];
            sS1b[i] = (bf16_t)v;
        }
        __syncthreads();
        float* a1 = sS0;                                   // 128x64 f32
        gemm_bf16(sS1b, 64, sW1, 64, a1, 64, nullptr, 128, 64, 64, wave, lane);
        __syncthreads();
        // shuffle1 -> A2 (256x32 bf16): rows (b*4+g)
        for (int i = tid; i < 8192; i += 256) {
            const int row = i >> 5, c = i & 31;
            const int b = row >> 2, g = row & 3;
            const int f = g * 32 + c;
            const int u = (f & 3) * 32 + (f >> 2);
            sS1b[i] = (bf16_t)a1[(2 * b + (u >> 6)) * 64 + (u & 63)];
        }
        __syncthreads();
        float* o2 = sS0;                                   // 256x32 f32
        gemm_bf16(sS1b, 32, sW2, 32, o2, 32, nullptr, 256, 32, 32, wave, lane);
        __syncthreads();
        // shuffle2 -> A3
        for (int i = tid; i < 8192; i += 256) {
            const int row = i >> 5, c = i & 31;
            const int b = row >> 2, g = row & 3;
            const int f = g * 32 + c;
            sS1b[i] = (bf16_t)o2[(b * 4 + (f & 3)) * 32 + (f >> 2)];
        }
        __syncthreads();
        float* o3 = sS0;                                   // 256x32 f32
        gemm_bf16(sS1b, 32, sW3, 32, o3, 32, nullptr, 256, 32, 32, wave, lane);
        __syncthreads();
        // relu + flatten -> A4 (64x128 bf16)
        for (int i = tid; i < 8192; i += 256) {
            const int b = i >> 7, u = i & 127;
            const float v = o3[(b * 4 + (u >> 5)) * 32 + (u & 31)];
            sS1b[i] = (bf16_t)(v > 0.0f ? v : 0.0f);
        }
        __syncthreads();
        float* o4 = sS0;                                   // 64x64 f32
        gemm_bf16(sS1b, 128, sW4, 128, o4, 64, nullptr, 64, 64, 128, wave, lane);
        __syncthreads();
        for (int i = tid; i < 4096; i += 256) sH[i] *= sigmf(o4[i]);
        __syncthreads();

        // ------------------- publish output --------------------------------
        if (layer == NLAYER - 1) {
            for (int i = tid; i < 4096; i += 256) {
                const int b = i >> 6, j = i & 63;
                out_ys[(size_t)b * 32768 + t * 64 + j] = sH[i];
            }
        } else {
            if (t >= NSLOT) {
                if (tid == 0) {
                    while (__hip_atomic_load(&done[layer], __ATOMIC_ACQUIRE,
                                             __HIP_MEMORY_SCOPE_AGENT)
                           < t - (NSLOT - 1))
                        __builtin_amdgcn_s_sleep(1);
                }
                __syncthreads();
            }
            float* slot =
                ring + ((size_t)layer * NSLOT + (t & (NSLOT - 1))) * 4096;
            for (int i = tid; i < 4096; i += 256) slot[i] = sH[i];
            __threadfence();
            __syncthreads();
            if (tid == 0)
                __hip_atomic_store(&ready[layer], t + 1, __ATOMIC_RELEASE,
                                   __HIP_MEMORY_SCOPE_AGENT);
        }
    }

    __syncthreads();
    for (int i = tid; i < 4096; i += 256)
        out_hfin[(size_t)layer * 4096 + i] = sH[i];
}

// ---------------------------------------------------------------------------
// Kernel 2: seg + cls heads. blockIdx.x = batch b; deterministic in-block
// tree reduction for cls (no float atomics).
// ---------------------------------------------------------------------------
__global__ void __launch_bounds__(256)
fss_heads(const float* __restrict__ ys,
          const float* __restrict__ sw1, const float* __restrict__ sb1,
          const float* __restrict__ sw2, const float* __restrict__ sb2,
          const float* __restrict__ sw3, const float* __restrict__ sb3,
          const float* __restrict__ sw4, const float* __restrict__ sb4,
          const float* __restrict__ cw1, const float* __restrict__ cb1,
          const float* __restrict__ cw2, const float* __restrict__ cb2,
          const float* __restrict__ cw3, const float* __restrict__ cb3,
          const float* __restrict__ cw4, const float* __restrict__ cb4,
          float* __restrict__ out_seg, float* __restrict__ out_cls) {
    __shared__ float red[256 * 4];
    const int b = blockIdx.x;
    const int tid = threadIdx.x;
    float acc[4] = {0.f, 0.f, 0.f, 0.f};

    for (int ch = 0; ch < 2; ++ch) {
        const int tw = ch * 256 + tid;
        const float* y = ys + ((size_t)b * LW + tw) * 64;
        float yv[64];
#pragma unroll
        for (int j = 0; j < 64; ++j) yv[j] = y[j];

        // ---- seg head ----
        float t1[8][4], s1[8][4], t2[8][4], s2[8][4], t3[8][4];
#pragma unroll
        for (int g = 0; g < 8; ++g)
#pragma unroll
            for (int s = 0; s < 4; ++s) {
                float a = sb1[s];
#pragma unroll
                for (int k = 0; k < 8; ++k) a += yv[g * 8 + k] * sw1[s * 8 + k];
                t1[g][s] = a;
            }
#pragma unroll
        for (int i = 0; i < 8; ++i)
#pragma unroll
            for (int j = 0; j < 4; ++j) {
                const int f = i * 4 + j;
                s1[i][j] = t1[f & 7][f >> 3];
            }
#pragma unroll
        for (int i = 0; i < 8; ++i)
#pragma unroll
            for (int s = 0; s < 4; ++s) {
                float a = sb2[s];
#pragma unroll
                for (int k = 0; k < 4; ++k) a += s1[i][k] * sw2[s * 4 + k];
                t2[i][s] = a;
            }
#pragma unroll
        for (int i = 0; i < 8; ++i)
#pragma unroll
            for (int j = 0; j < 4; ++j) {
                const int f = i * 4 + j;
                s2[i][j] = t2[f & 7][f >> 3];
            }
#pragma unroll
        for (int i = 0; i < 8; ++i)
#pragma unroll
            for (int s = 0; s < 4; ++s) {
                float a = sb3[s];
#pragma unroll
                for (int k = 0; k < 4; ++k) a += s2[i][k] * sw3[s * 4 + k];
                t3[i][s] = a;
            }
        // reshape (8,4)->(4,8), @ seg_w4^T (16x8), write transposed to seg
#pragma unroll
        for (int s = 0; s < 4; ++s) {
            float m[8];
#pragma unroll
            for (int g = 0; g < 8; ++g) {
                const int f = s * 8 + g;
                m[g] = t3[f >> 2][f & 3];
            }
#pragma unroll
            for (int w = 0; w < 16; ++w) {
                float a = sb4[w];
#pragma unroll
                for (int g = 0; g < 8; ++g) a += m[g] * sw4[w * 8 + g];
                out_seg[((size_t)b * 8192 + tw * 16 + w) * 4 + s] = a;
            }
        }

        // ---- cls head ----
        float u1[8][4], v1[8][4], u2[8][4], v2[8][4], u3[8][4];
#pragma unroll
        for (int g = 0; g < 8; ++g)
#pragma unroll
            for (int s = 0; s < 4; ++s) {
                float a = cb1[s];
#pragma unroll
                for (int k = 0; k < 8; ++k) a += yv[g * 8 + k] * cw1[s * 8 + k];
                u1[g][s] = a;
            }
#pragma unroll
        for (int i = 0; i < 8; ++i)
#pragma unroll
            for (int j = 0; j < 4; ++j) {
                const int f = i * 4 + j;
                v1[i][j] = u1[f & 7][f >> 3];
            }
#pragma unroll
        for (int i = 0; i < 8; ++i)
#pragma unroll
            for (int s = 0; s < 4; ++s) {
                float a = cb2[s];
#pragma unroll
                for (int k = 0; k < 4; ++k) a += v1[i][k] * cw2[s * 4 + k];
                u2[i][s] = a;
            }
#pragma unroll
        for (int i = 0; i < 8; ++i)
#pragma unroll
            for (int j = 0; j < 4; ++j) {
                const int f = i * 4 + j;
                v2[i][j] = u2[f & 7][f >> 3];
            }
#pragma unroll
        for (int i = 0; i < 8; ++i)
#pragma unroll
            for (int s = 0; s < 4; ++s) {
                float a = cb3[s];
#pragma unroll
                for (int k = 0; k < 4; ++k) a += v2[i][k] * cw3[s * 4 + k];
                u3[i][s] = (a > 0.f) ? a : 0.f;
            }
#pragma unroll
        for (int s = 0; s < 4; ++s) {
            float a = cb4[s];
#pragma unroll
            for (int q = 0; q < 32; ++q)
                a += u3[q >> 2][q & 3] * cw4[s * 32 + q];
            acc[s] += a;
        }
    }

#pragma unroll
    for (int s = 0; s < 4; ++s) red[tid * 4 + s] = acc[s];
    __syncthreads();
    for (int off = 128; off > 0; off >>= 1) {
        if (tid < off) {
#pragma unroll
            for (int s = 0; s < 4; ++s)
                red[tid * 4 + s] += red[(tid + off) * 4 + s];
        }
        __syncthreads();
    }
    if (tid < 4) out_cls[b * 4 + tid] = red[tid] / 512.0f;
}

// ---------------------------------------------------------------------------
// Kernel 3: 11-tap causal average pool over seg (window, not a scan).
// ---------------------------------------------------------------------------
__global__ void __launch_bounds__(256)
fss_pool(const float* __restrict__ seg, float* __restrict__ pool, int total) {
    const int idx = blockIdx.x * 256 + threadIdx.x;
    if (idx >= total) return;
    const int s = idx & 3;
    const int rest = idx >> 2;
    const int i = rest & 8191;
    const int b = rest >> 13;
    float sum = 0.0f;
    int lo = i - 10;
    if (lo < 0) lo = 0;
    for (int k = lo; k <= i; ++k)
        sum += seg[((size_t)b * 8192 + k) * 4 + s];
    pool[idx] = sum * (1.0f / 11.0f);
}

// ---------------------------------------------------------------------------
extern "C" void kernel_launch(void* const* d_in, const int* in_sizes, int n_in,
                              void* d_out, int out_size, void* d_ws,
                              size_t ws_size, hipStream_t stream) {
    (void)in_sizes; (void)n_in; (void)out_size; (void)ws_size;
    const float* x    = (const float*)d_in[0];
    const float* h0   = (const float*)d_in[1];
    const float* gwih = (const float*)d_in[2];
    const float* gwhh = (const float*)d_in[3];
    const float* gbih = (const float*)d_in[4];
    const float* gbhh = (const float*)d_in[5];
    const float* aw1  = (const float*)d_in[6];
    const float* aw2  = (const float*)d_in[7];
    const float* aw3  = (const float*)d_in[8];
    const float* aw4  = (const float*)d_in[9];
    const float* sw1  = (const float*)d_in[10];
    const float* sb1  = (const float*)d_in[11];
    const float* sw2  = (const float*)d_in[12];
    const float* sb2  = (const float*)d_in[13];
    const float* sw3  = (const float*)d_in[14];
    const float* sb3  = (const float*)d_in[15];
    const float* sw4  = (const float*)d_in[16];
    const float* sb4  = (const float*)d_in[17];
    const float* cw1  = (const float*)d_in[18];
    const float* cb1  = (const float*)d_in[19];
    const float* cw2  = (const float*)d_in[20];
    const float* cb2  = (const float*)d_in[21];
    const float* cw3  = (const float*)d_in[22];
    const float* cb3  = (const float*)d_in[23];
    const float* cw4  = (const float*)d_in[24];
    const float* cb4  = (const float*)d_in[25];

    float* out = (float*)d_out;
    float* out_pool = out;                 // (64, 8192, 4)
    float* out_cls  = out + 2097152;       // (64, 4)
    float* out_ys   = out + 2097408;       // (64, 512, 64)
    float* out_hf   = out + 4194560;       // (8, 64, 64)
    float* out_seg  = out + 4227328;       // (64, 8192, 4)

    int*   flags = (int*)d_ws;                       // ready[8], done[8]
    float* ring  = (float*)((char*)d_ws + 256);      // 7*4*4096 f32

    (void)hipFuncSetAttribute(
        reinterpret_cast<const void*>(fss_pipeline),
        hipFuncAttributeMaxDynamicSharedMemorySize, LDS_TOTAL);

    fss_init_flags<<<1, 64, 0, stream>>>(flags);
    fss_pipeline<<<NLAYER, 256, LDS_TOTAL, stream>>>(
        x, h0, gwih, gwhh, gbih, gbhh, aw1, aw2, aw3, aw4,
        out_ys, out_hf, flags, ring);
    fss_heads<<<BATCH, 256, 0, stream>>>(
        out_ys, sw1, sb1, sw2, sb2, sw3, sb3, sw4, sb4,
        cw1, cb1, cw2, cb2, cw3, cb3, cw4, cb4, out_seg, out_cls);
    fss_pool<<<(2097152 + 255) / 256, 256, 0, stream>>>(
        out_seg, out_pool, 2097152);
}
